// CustomMultiheadAttention_34935263985683
// MI455X (gfx1250) — compile-verified, matrix operand order checked by reading, not run on hardware
//
#include <hip/hip_runtime.h>
#include <hip/hip_bf16.h>

#define EDIM 1024
#define NH   16
#define HD   64
#define BSZ  4
#define SEQ  2048
#define BS   (BSZ*SEQ)      // 8192 tokens
#define SM_SCALE 0.125f     // 1/sqrt(64)

typedef __attribute__((ext_vector_type(16))) __bf16        v16bf;
typedef __attribute__((ext_vector_type(2)))  __bf16        v2bf;
typedef __attribute__((ext_vector_type(8)))  float         v8f;
typedef __attribute__((ext_vector_type(4)))  unsigned int  u32x4;
typedef __attribute__((ext_vector_type(2)))  unsigned int  u32x2;
typedef __attribute__((ext_vector_type(4)))  float         f32x4;

union Frag16 { v16bf v; u32x4 q[2]; };

// --- f32 -> bf16 conversion -------------------------------------------------
// Plain fptrunc casts: the backend picks the native converter if gfx1250 has
// one (v_cvt_pk_bf16_f32 family), else expands to the standard RNE sequence.
__device__ __forceinline__ unsigned pk2_bf16(float a, float b) {
  union { v2bf v; unsigned u; } c;
  c.v.x = (__bf16)a;
  c.v.y = (__bf16)b;
  return c.u;
}

__device__ __forceinline__ unsigned short f2bf(float f) {
  union { __bf16 b; unsigned short u; } c;
  c.b = (__bf16)f;
  return c.u;
}

__device__ __forceinline__ u32x2 pack4_bf16(f32x4 v) {
  u32x2 r;
  r.x = pk2_bf16(v.x, v.y);
  r.y = pk2_bf16(v.z, v.w);
  return r;
}

__device__ __forceinline__ v8f wmma_bf16(v16bf a, v16bf b, v8f c) {
  // (neg_a, A, neg_b, B, c_mod, C, reuse_a, reuse_b)
  return __builtin_amdgcn_wmma_f32_16x16x32_bf16(false, a, false, b, (short)0, c, false, false);
}

// gfx1250 async global->LDS copy (ASYNCcnt path); per-lane 16B transfer.
// lds_off = wave-relative LDS byte offset (low 32 bits of generic shared addr).
__device__ __forceinline__ void async_b128(unsigned lds_off, const void* gaddr) {
  asm volatile("global_load_async_to_lds_b128 %0, %1, off"
               :: "v"(lds_off), "v"((unsigned long long)(size_t)gaddr)
               : "memory");
}
__device__ __forceinline__ void wait_async0() {
  asm volatile("s_wait_asynccnt 0x0" ::: "memory");
}

// ---------------------------------------------------------------------------
// Kernel 1: QKV projection.  y = x @ W + b, one of {q,k,v} per blockIdx.z.
// 128x128 tile per block (8 wave32 -> 64x32 each), K-step = 32.
// Output written bf16 in [B][H][S][D] layout for attention.
// ---------------------------------------------------------------------------
__global__ __launch_bounds__(256)
void qkv_proj_kernel(const float* __restrict__ x,
                     const float* __restrict__ Wq, const float* __restrict__ bq,
                     const float* __restrict__ Wk, const float* __restrict__ bk,
                     const float* __restrict__ Wv, const float* __restrict__ bv,
                     unsigned short* __restrict__ qkv)
{
  constexpr int LDA = 40, LDB = 40;                  // pad: 40*2B = 80B rows (16B aligned)
  __shared__ unsigned short As[128 * LDA];           // x tile   [m][k] bf16
  __shared__ unsigned short Bt[128 * LDB];           // W tile^T [n][k] bf16

  const int z = blockIdx.z;
  const float* W    = (z == 0) ? Wq : (z == 1) ? Wk : Wv;
  const float* bias = (z == 0) ? bq : (z == 1) ? bk : bv;
  unsigned short* outp = qkv + (size_t)z * BS * EDIM;

  const int tid  = threadIdx.x;
  const int wave = tid >> 5, lane = tid & 31;
  const int g = lane >> 4, nl = lane & 15;
  const int blockM = blockIdx.x * 128;
  const int blockN = blockIdx.y * 128;
  const int wm = (wave & 1) * 64;
  const int wn = (wave >> 1) * 32;

  v8f acc[4][2];
  for (int i = 0; i < 4; ++i)
    for (int j = 0; j < 2; ++j)
      for (int e = 0; e < 8; ++e) acc[i][j][e] = 0.0f;

  for (int kt = 0; kt < EDIM / 32; ++kt) {
    // stage A: 128x32 f32 -> bf16, 128b global loads + 64b LDS stores
    #pragma unroll
    for (int j = 0; j < 4; ++j) {
      int i = tid + j * 256;
      int r = i >> 3, c4 = (i & 7) * 4;
      f32x4 v = *reinterpret_cast<const f32x4*>(
          &x[(size_t)(blockM + r) * EDIM + kt * 32 + c4]);
      *reinterpret_cast<u32x2*>(&As[r * LDA + c4]) = pack4_bf16(v);
    }
    // stage B transposed: 128b global loads, b16/b16_d16_hi scatter stores
    #pragma unroll
    for (int j = 0; j < 4; ++j) {
      int i = tid + j * 256;
      int k = i >> 5, n4 = (i & 31) * 4;
      f32x4 v = *reinterpret_cast<const f32x4*>(
          &W[(size_t)(kt * 32 + k) * EDIM + blockN + n4]);
      unsigned p01 = pk2_bf16(v.x, v.y);
      unsigned p23 = pk2_bf16(v.z, v.w);
      Bt[(n4 + 0) * LDB + k] = (unsigned short)p01;
      Bt[(n4 + 1) * LDB + k] = (unsigned short)(p01 >> 16);
      Bt[(n4 + 2) * LDB + k] = (unsigned short)p23;
      Bt[(n4 + 3) * LDB + k] = (unsigned short)(p23 >> 16);
    }
    // prefetch next K-step tiles toward the WGP while WMMAs run
    if (kt + 1 < EDIM / 32) {
      __builtin_prefetch(&x[(size_t)(blockM + (tid >> 1)) * EDIM + (kt + 1) * 32 + (tid & 1) * 16], 0, 3);
      __builtin_prefetch(&W[(size_t)((kt + 1) * 32 + (tid >> 3)) * EDIM + blockN + (tid & 7) * 16], 0, 3);
    }
    __syncthreads();

    Frag16 a[4], b[2];
    #pragma unroll
    for (int fm = 0; fm < 4; ++fm) {                 // A: lane=row M, K=(e&7)+16*(e>=8)+8g
      const u32x4* p = reinterpret_cast<const u32x4*>(&As[(wm + fm * 16 + nl) * LDA + g * 8]);
      a[fm].q[0] = p[0]; a[fm].q[1] = p[2];
    }
    #pragma unroll
    for (int fn = 0; fn < 2; ++fn) {                 // B from transposed tile: same chunking
      const u32x4* p = reinterpret_cast<const u32x4*>(&Bt[(wn + fn * 16 + nl) * LDB + g * 8]);
      b[fn].q[0] = p[0]; b[fn].q[1] = p[2];
    }
    #pragma unroll
    for (int fm = 0; fm < 4; ++fm)
      #pragma unroll
      for (int fn = 0; fn < 2; ++fn)
        acc[fm][fn] = wmma_bf16(a[fm].v, b[fn].v, acc[fm][fn]);
    __syncthreads();
  }

  // epilogue: bias, bf16 store into [B][H][S][D]
  #pragma unroll
  for (int fm = 0; fm < 4; ++fm)
    #pragma unroll
    for (int fn = 0; fn < 2; ++fn) {
      int c = blockN + wn + fn * 16 + nl;
      float bb = bias[c];
      int h = c >> 6, d = c & 63;
      #pragma unroll
      for (int r = 0; r < 8; ++r) {
        int t = blockM + wm + fm * 16 + r + 8 * g;   // C layout: M = r + 8*lanegroup
        int bIdx = t >> 11, s = t & 2047;
        outp[(((size_t)bIdx * NH + h) * SEQ + s) * HD + d] = f2bf(acc[fm][fn][r] + bb);
      }
    }
}

// ---------------------------------------------------------------------------
// Kernel 2: flash attention, bf16 inputs, fp32 online softmax.
// Block = 128 threads = 4 waves; each wave owns 16 query rows; q-tile = 64.
// K tile staged via async global->LDS DMA; V staged transposed via VGPRs.
// 16 WMMAs per wave per 64-key tile.
// ---------------------------------------------------------------------------
__global__ __launch_bounds__(128)
void flash_attn_kernel(const unsigned short* __restrict__ Qb,
                       const unsigned short* __restrict__ Kb,
                       const unsigned short* __restrict__ Vb,
                       unsigned short* __restrict__ Ob)   // [B][S][H*D] bf16
{
  constexpr int LK = 72;                              // 144B rows, 16B aligned
  __shared__ unsigned short Ks[64 * LK];              // K tile [key][d]
  __shared__ unsigned short Vt[64 * LK];              // V tile transposed [d][key]
  __shared__ unsigned short Ps[4 * 16 * LK];          // per-wave P tile [row][key]

  const int tid = threadIdx.x, wave = tid >> 5, lane = tid & 31;
  const int g = lane >> 4, nl = lane & 15;
  const int h = blockIdx.y, b = blockIdx.z;
  const int q0 = blockIdx.x * 64 + wave * 16;
  const size_t bh = (size_t)b * NH + h;
  const unsigned short* qrow = Qb + bh * SEQ * HD;
  const unsigned short* krow = Kb + bh * SEQ * HD;
  const unsigned short* vrow = Vb + bh * SEQ * HD;
  const unsigned ks_base = (unsigned)(size_t)(void*)Ks;  // LDS byte offset

  // Q fragments: 16 rows x 64 d -> two 16x32 A-frags, loaded once from global
  Frag16 qf[2];
  {
    const u32x4* qp = reinterpret_cast<const u32x4*>(qrow + (size_t)(q0 + nl) * HD);
    qf[0].q[0] = qp[g];     qf[0].q[1] = qp[g + 2];   // d 0..31
    qf[1].q[0] = qp[4 + g]; qf[1].q[1] = qp[6 + g];   // d 32..63
  }

  float Mx[8], L[8], alpha[8];
  v8f Oacc[4];
  for (int r = 0; r < 8; ++r) { Mx[r] = -__builtin_inff(); L[r] = 0.0f; }
  for (int a = 0; a < 4; ++a)
    for (int r = 0; r < 8; ++r) Oacc[a][r] = 0.0f;

  for (int kt = 0; kt < SEQ / 64; ++kt) {
    const unsigned short* kg = krow + (size_t)kt * 64 * HD;
    const unsigned short* vg = vrow + (size_t)kt * 64 * HD;
    #pragma unroll
    for (int j = 0; j < 4; ++j) {                     // 64 rows * 8 16B-chunks
      int i = tid + j * 128;
      int r = i >> 3, c = i & 7;
      // K: async DMA straight into LDS (no VGPR round trip)
      async_b128(ks_base + (unsigned)(r * LK + c * 8) * 2u,
                 kg + (size_t)r * HD + c * 8);
      // V: VGPR transpose into LDS
      u32x4 vv = *reinterpret_cast<const u32x4*>(vg + (size_t)r * HD + c * 8);
      const unsigned short* e = reinterpret_cast<const unsigned short*>(&vv);
      #pragma unroll
      for (int jj = 0; jj < 8; ++jj) Vt[(c * 8 + jj) * LK + r] = e[jj];
    }
    wait_async0();
    __syncthreads();

    // S = Q K^T  (matmul K-dim = d; B-frag lane = key row, contiguous d chunks)
    v8f Sacc[4];
    for (int a = 0; a < 4; ++a)
      for (int r = 0; r < 8; ++r) Sacc[a][r] = 0.0f;
    #pragma unroll
    for (int a = 0; a < 4; ++a)
      #pragma unroll
      for (int ds = 0; ds < 2; ++ds) {
        Frag16 bf;
        const u32x4* p = reinterpret_cast<const u32x4*>(&Ks[(a * 16 + nl) * LK + ds * 32 + g * 8]);
        bf.q[0] = p[0]; bf.q[1] = p[2];
        Sacc[a] = wmma_bf16(qf[ds].v, bf.v, Sacc[a]);
      }

    // online softmax (row = r + 8g; row spans the 16 lanes of each half-wave)
    #pragma unroll
    for (int a = 0; a < 4; ++a)
      for (int r = 0; r < 8; ++r) Sacc[a][r] *= SM_SCALE;
    #pragma unroll
    for (int r = 0; r < 8; ++r) {
      float m = fmaxf(fmaxf(Sacc[0][r], Sacc[1][r]), fmaxf(Sacc[2][r], Sacc[3][r]));
      m = fmaxf(m, __shfl_xor(m, 1));
      m = fmaxf(m, __shfl_xor(m, 2));
      m = fmaxf(m, __shfl_xor(m, 4));
      m = fmaxf(m, __shfl_xor(m, 8));
      float mn = fmaxf(Mx[r], m);
      alpha[r] = __expf(Mx[r] - mn);                  // first iter: exp(-inf)=0
      Mx[r] = mn;
      L[r] *= alpha[r];
    }
    unsigned short* pw = &Ps[wave * 16 * LK];
    #pragma unroll
    for (int a = 0; a < 4; ++a)
      #pragma unroll
      for (int r = 0; r < 8; ++r) {
        float p = __expf(Sacc[a][r] - Mx[r]);
        L[r] += p;                                    // per-lane partial row sum
        pw[(r + 8 * g) * LK + a * 16 + nl] = f2bf(p); // C-layout -> A-layout via LDS
      }
    #pragma unroll
    for (int a = 0; a < 4; ++a)
      for (int r = 0; r < 8; ++r) Oacc[a][r] *= alpha[r];

    // O += P V  (matmul K-dim = keys; B-frags from transposed V tile)
    Frag16 pf[2];
    {
      const u32x4* p0 = reinterpret_cast<const u32x4*>(pw + nl * LK + g * 8);
      pf[0].q[0] = p0[0]; pf[0].q[1] = p0[2];         // keys 0..31
      const u32x4* p1 = reinterpret_cast<const u32x4*>(pw + nl * LK + 32 + g * 8);
      pf[1].q[0] = p1[0]; pf[1].q[1] = p1[2];         // keys 32..63
    }
    #pragma unroll
    for (int a = 0; a < 4; ++a)                        // d-frag
      #pragma unroll
      for (int ks = 0; ks < 2; ++ks) {
        Frag16 bf;
        const u32x4* p = reinterpret_cast<const u32x4*>(&Vt[(a * 16 + nl) * LK + ks * 32 + g * 8]);
        bf.q[0] = p[0]; bf.q[1] = p[2];
        Oacc[a] = wmma_bf16(pf[ks].v, bf.v, Oacc[a]);
      }
    __syncthreads();
  }

  // finalize: reduce row sums over 16 lanes, normalize, store bf16 [B][S][E]
  #pragma unroll
  for (int r = 0; r < 8; ++r) {
    float s = L[r];
    s += __shfl_xor(s, 1); s += __shfl_xor(s, 2);
    s += __shfl_xor(s, 4); s += __shfl_xor(s, 8);
    L[r] = 1.0f / s;
  }
  #pragma unroll
  for (int a = 0; a < 4; ++a)
    #pragma unroll
    for (int r = 0; r < 8; ++r) {
      int srow = q0 + r + 8 * g;
      int col  = h * HD + a * 16 + nl;
      Ob[((size_t)b * SEQ + srow) * EDIM + col] = f2bf(Oacc[a][r] * L[r]);
    }
}

// ---------------------------------------------------------------------------
// Kernel 3: output projection.  out = O @ Wo + bo  (bf16 A via async DMA)
// ---------------------------------------------------------------------------
__global__ __launch_bounds__(256)
void out_proj_kernel(const unsigned short* __restrict__ Ob,
                     const float* __restrict__ Wo, const float* __restrict__ bo,
                     float* __restrict__ out)
{
  constexpr int LDA = 40, LDB = 40;
  __shared__ unsigned short As[128 * LDA];
  __shared__ unsigned short Bt[128 * LDB];

  const int tid  = threadIdx.x;
  const int wave = tid >> 5, lane = tid & 31;
  const int g = lane >> 4, nl = lane & 15;
  const int blockM = blockIdx.x * 128;
  const int blockN = blockIdx.y * 128;
  const int wm = (wave & 1) * 64;
  const int wn = (wave >> 1) * 32;
  const unsigned as_base = (unsigned)(size_t)(void*)As;

  v8f acc[4][2];
  for (int i = 0; i < 4; ++i)
    for (int j = 0; j < 2; ++j)
      for (int e = 0; e < 8; ++e) acc[i][j][e] = 0.0f;

  for (int kt = 0; kt < EDIM / 32; ++kt) {
    // A tile already bf16: async DMA global->LDS (512 x 16B chunks)
    #pragma unroll
    for (int j = 0; j < 2; ++j) {
      int i = tid + j * 256;
      int r = i >> 2, c = (i & 3) * 8;
      async_b128(as_base + (unsigned)(r * LDA + c) * 2u,
                 Ob + (size_t)(blockM + r) * EDIM + kt * 32 + c);
    }
    // B tile: f32 -> bf16 transposed scatter
    #pragma unroll
    for (int j = 0; j < 4; ++j) {
      int i = tid + j * 256;
      int k = i >> 5, n4 = (i & 31) * 4;
      f32x4 v = *reinterpret_cast<const f32x4*>(
          &Wo[(size_t)(kt * 32 + k) * EDIM + blockN + n4]);
      unsigned p01 = pk2_bf16(v.x, v.y);
      unsigned p23 = pk2_bf16(v.z, v.w);
      Bt[(n4 + 0) * LDB + k] = (unsigned short)p01;
      Bt[(n4 + 1) * LDB + k] = (unsigned short)(p01 >> 16);
      Bt[(n4 + 2) * LDB + k] = (unsigned short)p23;
      Bt[(n4 + 3) * LDB + k] = (unsigned short)(p23 >> 16);
    }
    wait_async0();
    __syncthreads();

    Frag16 a[4], b[2];
    #pragma unroll
    for (int fm = 0; fm < 4; ++fm) {
      const u32x4* p = reinterpret_cast<const u32x4*>(&As[(wm + fm * 16 + nl) * LDA + g * 8]);
      a[fm].q[0] = p[0]; a[fm].q[1] = p[2];
    }
    #pragma unroll
    for (int fn = 0; fn < 2; ++fn) {
      const u32x4* p = reinterpret_cast<const u32x4*>(&Bt[(wn + fn * 16 + nl) * LDB + g * 8]);
      b[fn].q[0] = p[0]; b[fn].q[1] = p[2];
    }
    #pragma unroll
    for (int fm = 0; fm < 4; ++fm)
      #pragma unroll
      for (int fn = 0; fn < 2; ++fn)
        acc[fm][fn] = wmma_bf16(a[fm].v, b[fn].v, acc[fm][fn]);
    __syncthreads();
  }

  #pragma unroll
  for (int fm = 0; fm < 4; ++fm)
    #pragma unroll
    for (int fn = 0; fn < 2; ++fn) {
      int c = blockN + wn + fn * 16 + nl;
      float bb = bo[c];
      #pragma unroll
      for (int r = 0; r < 8; ++r) {
        int t = blockM + wm + fm * 16 + r + 8 * g;
        out[(size_t)t * EDIM + c] = acc[fm][fn][r] + bb;
      }
    }
}

// ---------------------------------------------------------------------------
extern "C" void kernel_launch(void* const* d_in, const int* in_sizes, int n_in,
                              void* d_out, int out_size, void* d_ws, size_t ws_size,
                              hipStream_t stream) {
  const float* x  = (const float*)d_in[0];
  const float* Wq = (const float*)d_in[1];
  const float* bq = (const float*)d_in[2];
  const float* Wk = (const float*)d_in[3];
  const float* bk = (const float*)d_in[4];
  const float* Wv = (const float*)d_in[5];
  const float* bv = (const float*)d_in[6];
  const float* Wo = (const float*)d_in[7];
  const float* bo = (const float*)d_in[8];
  float* out = (float*)d_out;

  unsigned short* qkv = (unsigned short*)d_ws;                 // 3 * BS*E bf16 = 48 MiB
  unsigned short* Ob  = qkv + (size_t)3 * BS * EDIM;           // BS*E bf16     = 16 MiB

  dim3 g1(BS / 128, EDIM / 128, 3);
  qkv_proj_kernel<<<g1, 256, 0, stream>>>(x, Wq, bq, Wk, bk, Wv, bv, qkv);

  dim3 g2(SEQ / 64, NH, BSZ);
  flash_attn_kernel<<<g2, 128, 0, stream>>>(qkv,
                                            qkv + (size_t)BS * EDIM,
                                            qkv + (size_t)2 * BS * EDIM, Ob);

  dim3 g3(BS / 128, EDIM / 128, 1);
  out_proj_kernel<<<g3, 256, 0, stream>>>(Ob, Wo, bo, out);
}
